// SimplifiedVarianceAdaptor_68298569941378
// MI455X (gfx1250) — matrix operand order repaired and checked by reading, not compiled, not run on hardware
//
#include <hip/hip_runtime.h>
#include <hip/hip_bf16.h>

// ---------------------------------------------------------------------------
// SimplifiedVarianceAdaptor for MI455X (gfx1250, wave32, WMMA, async-LDS)
// B=16, S=1024, D=256, F=256, T=8192, K=3
//
// Memory-bound workload (~330MB HBM traffic vs ~41 GFLOP):
//  * all GEMMs run v_wmma_f32_16x16x32_bf16 (bf16 in, f32 accumulate)
//  * 4 M-tiles per wave share one B fragment (4x WMMA density)
//  * conv weights staged global->LDS with global_load_async_to_lds_b128
//    (+ s_wait_asynccnt), B fragments then read from LDS (bank-conflict-free
//    via row padding to 772 floats)
//  * A-tile loads vectorized to b128; streaming rows prefetched
// ---------------------------------------------------------------------------

typedef __attribute__((ext_vector_type(16))) __bf16 v16bf;
typedef __attribute__((ext_vector_type(8)))  float  v8f;

#define BATCH   16
#define SEQ     1024
#define DIM     256
#define FDIM    256
#define TMAX    8192
#define NROWS   (BATCH * SEQ)      // 16384
#define WPAD    772                // padded LDS row stride (floats): 16B aligned,
                                   // 772 % 64 == 4 -> 16 lanes hit 16 distinct banks

__device__ __forceinline__ __bf16 tobf(float f) { return (__bf16)f; }

// ---------------------------------------------------------------------------
// Init reduction slots. red[0]=max||xn|| (0), red[1]=min||x|| (+FLT_MAX bits),
// red[2]=max||x|| (0).  Positive floats -> uint bit ordering matches.
// ---------------------------------------------------------------------------
__global__ void sva_init_kernel(unsigned int* red) {
  red[0] = 0u;
  red[1] = 0x7F7FFFFFu;
  red[2] = 0u;
}

// ---------------------------------------------------------------------------
// XP = x + note_pitch @ np_w + np_b
// One wave per block; each wave owns 4 M-tiles x 1 N-tile (64x16 output).
// grid = (DIM/16, NROWS/64), block = 32
// ---------------------------------------------------------------------------
__global__ __launch_bounds__(32)
void sva_noteproj_kernel(const float* __restrict__ notep,
                         const float* __restrict__ w,      // [D,D] (cin,cout)
                         const float* __restrict__ bias,   // [D]
                         const float* __restrict__ xin,    // [NROWS, D]
                         float* __restrict__ out) {        // [NROWS, D]
  const int lane  = threadIdx.x & 31;
  const int half  = lane >> 4;
  const int l15   = lane & 15;
  const int tileN = blockIdx.x * 16;
  const int tileM0 = blockIdx.y * 64;
  const int n = tileN + l15;

  // prefetch the 4 streaming A rows + epilogue rows (global_prefetch_b8)
#pragma unroll
  for (int t = 0; t < 4; ++t) {
    __builtin_prefetch(notep + (size_t)(tileM0 + 16 * t + l15) * DIM, 0, 1);
    __builtin_prefetch(xin   + (size_t)(tileM0 + 16 * t + l15) * DIM + tileN, 0, 1);
  }

  v8f acc[4] = {};
  for (int k0 = 0; k0 < DIM; k0 += 32) {
    // B fragment (shared across the 4 M-tiles)
    v16bf bfr;
#pragma unroll
    for (int v = 0; v < 8; ++v) {
      const int cc = k0 + 8 * half + 2 * v + 8 * (v >> 2);
      bfr[2 * v]     = tobf(w[(size_t)cc * DIM + n]);
      bfr[2 * v + 1] = tobf(w[(size_t)(cc + 1) * DIM + n]);
    }
#pragma unroll
    for (int t = 0; t < 4; ++t) {
      const int m = tileM0 + 16 * t + l15;
      const float4* ap = (const float4*)(notep + (size_t)m * DIM + k0 + 8 * half);
      const float4 a0 = ap[0], a1 = ap[1];   // K = 8*half + 0..7
      const float4 a2 = ap[4], a3 = ap[5];   // K = 16 + 8*half + 0..7
      v16bf a;
      a[0]  = tobf(a0.x); a[1]  = tobf(a0.y); a[2]  = tobf(a0.z); a[3]  = tobf(a0.w);
      a[4]  = tobf(a1.x); a[5]  = tobf(a1.y); a[6]  = tobf(a1.z); a[7]  = tobf(a1.w);
      a[8]  = tobf(a2.x); a[9]  = tobf(a2.y); a[10] = tobf(a2.z); a[11] = tobf(a2.w);
      a[12] = tobf(a3.x); a[13] = tobf(a3.y); a[14] = tobf(a3.z); a[15] = tobf(a3.w);
      acc[t] = __builtin_amdgcn_wmma_f32_16x16x32_bf16(false, a, false, bfr,
                                                       (short)0, acc[t], false, false);
    }
  }
#pragma unroll
  for (int t = 0; t < 4; ++t) {
#pragma unroll
    for (int r = 0; r < 8; ++r) {
      const int row = tileM0 + 16 * t + r + 8 * half;
      const int col = tileN + l15;
      const size_t o = (size_t)row * DIM + col;
      out[o] = acc[t][r] + xin[o] + bias[col];
    }
  }
}

// ---------------------------------------------------------------------------
// LayerNorm over D + row norms + global min/max atomics.  256 threads / row.
// ---------------------------------------------------------------------------
__global__ __launch_bounds__(256)
void sva_ln_norm_kernel(const float* __restrict__ xp,
                        const float* __restrict__ g,
                        const float* __restrict__ bb,
                        float* __restrict__ xn,
                        float* __restrict__ normXN,
                        float* __restrict__ normXP,
                        unsigned int* __restrict__ red) {
  __shared__ float s1[256];
  __shared__ float s2[256];
  const int row = blockIdx.x;
  const int c = threadIdx.x;
  const float v = xp[(size_t)row * DIM + c];
  s1[c] = v;
  s2[c] = v * v;
  __syncthreads();
  for (int off = 128; off > 0; off >>= 1) {
    if (c < off) { s1[c] += s1[c + off]; s2[c] += s2[c + off]; }
    __syncthreads();
  }
  const float mean  = s1[0] * (1.0f / DIM);
  const float sumsq = s2[0];
  const float var   = sumsq * (1.0f / DIM) - mean * mean;
  const float rstd  = rsqrtf(var + 1e-5f);
  const float y = g[c] * (v - mean) * rstd + bb[c];
  xn[(size_t)row * DIM + c] = y;
  __syncthreads();
  s1[c] = y * y;
  __syncthreads();
  for (int off = 128; off > 0; off >>= 1) {
    if (c < off) s1[c] += s1[c + off];
    __syncthreads();
  }
  if (c == 0) {
    const float nXN = sqrtf(s1[0]);
    const float nXP = sqrtf(sumsq);
    normXN[row] = nXN;
    normXP[row] = nXP;
    atomicMax(&red[0], __float_as_uint(nXN));
    atomicMin(&red[1], __float_as_uint(nXP));
    atomicMax(&red[2], __float_as_uint(nXP));
  }
}

// ---------------------------------------------------------------------------
// Hout = relu(conv1d(Hin, w[O,C,3], SAME) + bias) as GEMM.
// Block = 128 threads (4 waves); block owns one 16-channel N-tile and 256 rows
// (each wave: 4 M-tiles).  The block's 16x768 weight slab is staged into LDS
// via global_load_async_to_lds_b128 and read back as B fragments via ds loads.
// grid = (FDIM/16, NROWS/256), block = 128
// ---------------------------------------------------------------------------
__global__ __launch_bounds__(128)
void sva_convgemm_kernel(const float* __restrict__ hin,
                         const float* __restrict__ w,     // [O, C, 3]
                         const float* __restrict__ bias,  // [O]
                         float* __restrict__ hout) {
  __shared__ __align__(16) float wsh[16 * WPAD];   // 49,408 B

  const int tid   = threadIdx.x;
  const int lane  = tid & 31;
  const int half  = lane >> 4;
  const int l15   = lane & 15;
  const int wave  = tid >> 5;
  const int tileN = blockIdx.x * 16;
  const int waveM = blockIdx.y * 256 + wave * 64;
  const int n = tileN + l15;

  // ---- async stage: 16 rows x 768 floats (192 x b128 chunks per row) ----
  {
    const unsigned int lds_base = (unsigned int)(size_t)(&wsh[0]);
    for (int c = tid; c < 16 * 192; c += 128) {
      const int r = c / 192;
      const int q = c - r * 192;
      const unsigned int loff = lds_base + (unsigned int)((r * WPAD + q * 4) * 4);
      const float* gsrc = w + (size_t)(tileN + r) * (FDIM * 3) + q * 4;
      asm volatile("global_load_async_to_lds_b128 %0, %1, off"
                   :: "v"(loff), "v"(gsrc) : "memory");
    }
    asm volatile("s_wait_asynccnt 0x0" ::: "memory");
    __syncthreads();
  }

  const float* wrow = &wsh[l15 * WPAD];   // this lane's output-channel row
  v8f acc[4] = {};

  for (int tap = 0; tap < 3; ++tap) {
    int   srcRow[4];
    float sc[4];
#pragma unroll
    for (int t = 0; t < 4; ++t) {
      const int m = waveM + 16 * t + l15;
      const int ss = (m & (SEQ - 1)) + tap - 1;
      const bool valid = (ss >= 0) && (ss < SEQ);
      srcRow[t] = valid ? (m + tap - 1) : m;   // clamp: always in-bounds
      sc[t]     = valid ? 1.0f : 0.0f;         // zero contribution at edges
    }
    for (int c0 = 0; c0 < FDIM; c0 += 32) {
      // B fragment from LDS (bank-conflict-free: lane stride WPAD)
      v16bf bfr;
#pragma unroll
      for (int v = 0; v < 8; ++v) {
        const int cc = c0 + 8 * half + 2 * v + 8 * (v >> 2);
        bfr[2 * v]     = tobf(wrow[cc * 3 + tap]);
        bfr[2 * v + 1] = tobf(wrow[(cc + 1) * 3 + tap]);
      }
#pragma unroll
      for (int t = 0; t < 4; ++t) {
        const float4* ap = (const float4*)(hin + (size_t)srcRow[t] * FDIM + c0 + 8 * half);
        const float s = sc[t];
        const float4 a0 = ap[0], a1 = ap[1];
        const float4 a2 = ap[4], a3 = ap[5];
        v16bf a;
        a[0]  = tobf(a0.x * s); a[1]  = tobf(a0.y * s); a[2]  = tobf(a0.z * s); a[3]  = tobf(a0.w * s);
        a[4]  = tobf(a1.x * s); a[5]  = tobf(a1.y * s); a[6]  = tobf(a1.z * s); a[7]  = tobf(a1.w * s);
        a[8]  = tobf(a2.x * s); a[9]  = tobf(a2.y * s); a[10] = tobf(a2.z * s); a[11] = tobf(a2.w * s);
        a[12] = tobf(a3.x * s); a[13] = tobf(a3.y * s); a[14] = tobf(a3.z * s); a[15] = tobf(a3.w * s);
        acc[t] = __builtin_amdgcn_wmma_f32_16x16x32_bf16(false, a, false, bfr,
                                                         (short)0, acc[t], false, false);
      }
    }
  }
#pragma unroll
  for (int t = 0; t < 4; ++t) {
#pragma unroll
    for (int r = 0; r < 8; ++r) {
      const int row = waveM + 16 * t + r + 8 * half;
      const int col = tileN + l15;
      const float v = acc[t][r] + bias[col];
      hout[(size_t)row * FDIM + col] = fmaxf(v, 0.0f);
    }
  }
}

// ---------------------------------------------------------------------------
// Duration-predictor final linear + epilogue.  One wave per row.
// ---------------------------------------------------------------------------
__global__ __launch_bounds__(32)
void sva_dp_final_kernel(const float* __restrict__ h2,
                         const float* __restrict__ wl,   // [F,1]
                         const float* __restrict__ bl,   // [1]
                         const unsigned char* __restrict__ mask,
                         const float* __restrict__ normXN,
                         const unsigned int* __restrict__ red,
                         float* __restrict__ ldur,
                         float* __restrict__ dur,
                         float* __restrict__ durWS) {
  const int row = blockIdx.x;
  const int lane = threadIdx.x;
  const float* h = h2 + (size_t)row * FDIM;
  float sum = 0.0f;
  for (int i = lane; i < FDIM; i += 32) sum += h[i] * wl[i];
  for (int off = 16; off > 0; off >>= 1) sum += __shfl_xor(sum, off, 32);
  if (lane == 0) {
    float base = sum + bl[0];
    if (mask[row]) base = 0.0f;
    const float maxn = __uint_as_float(red[0]);
    const float escale = 0.8f + 0.4f * (normXN[row] / maxn);
    const int s = row & (SEQ - 1);
    const float pscale = 1.0f + 0.1f * ((float)s / (float)SEQ);
    float ld = base * escale * pscale;
    if (mask[row]) ld = 0.0f;
    const float d = expf(ld);   // tempo_factor = 1.0
    ldur[row]  = ld;
    dur[row]   = d;
    durWS[row] = d;
  }
}

// ---------------------------------------------------------------------------
// Pitch-predictor final linear + f0 modification.  One wave per row.
// ---------------------------------------------------------------------------
__global__ __launch_bounds__(32)
void sva_pp_final_kernel(const float* __restrict__ h2,
                         const float* __restrict__ wl,   // [F,3]
                         const float* __restrict__ bl,   // [3]
                         const unsigned char* __restrict__ mask,
                         const float* __restrict__ normXP,
                         const unsigned int* __restrict__ red,
                         float* __restrict__ pout) {
  const int row = blockIdx.x;
  const int lane = threadIdx.x;
  const float* h = h2 + (size_t)row * FDIM;
  float s0 = 0.0f, s1 = 0.0f, s2 = 0.0f;
  for (int i = lane; i < FDIM; i += 32) {
    const float hv = h[i];
    s0 += hv * wl[i * 3 + 0];
    s1 += hv * wl[i * 3 + 1];
    s2 += hv * wl[i * 3 + 2];
  }
  for (int off = 16; off > 0; off >>= 1) {
    s0 += __shfl_xor(s0, off, 32);
    s1 += __shfl_xor(s1, off, 32);
    s2 += __shfl_xor(s2, off, 32);
  }
  if (lane == 0) {
    float p0 = s0 + bl[0], p1 = s1 + bl[1], p2 = s2 + bl[2];
    if (mask[row]) { p0 = 0.0f; p1 = 0.0f; p2 = 0.0f; }
    const float mn = __uint_as_float(red[1]);
    const float mx = __uint_as_float(red[2]);
    const float en = (normXP[row] - mn) / (mx - mn + 1e-8f);
    const float f0s = 100.0f + 400.0f * en;
    const float f0b = expf(p0) * f0s / 220.0f;
    const float mf0 = logf(f0b + 1e-8f);
    pout[(size_t)row * 3 + 0] = mf0;
    pout[(size_t)row * 3 + 1] = p1;
    pout[(size_t)row * 3 + 2] = p2;
  }
}

// ---------------------------------------------------------------------------
// Energy-predictor final linear.  One wave per row.
// ---------------------------------------------------------------------------
__global__ __launch_bounds__(32)
void sva_ep_final_kernel(const float* __restrict__ h2,
                         const float* __restrict__ wl,
                         const float* __restrict__ bl,
                         const unsigned char* __restrict__ mask,
                         float* __restrict__ eout) {
  const int row = blockIdx.x;
  const int lane = threadIdx.x;
  const float* h = h2 + (size_t)row * FDIM;
  float sum = 0.0f;
  for (int i = lane; i < FDIM; i += 32) sum += h[i] * wl[i];
  for (int off = 16; off > 0; off >>= 1) sum += __shfl_xor(sum, off, 32);
  if (lane == 0) {
    float v = sum + bl[0];
    if (mask[row]) v = 0.0f;
    eout[row] = v;
  }
}

// ---------------------------------------------------------------------------
// Per-batch inclusive cumsum of rounded durations, clipped at TMAX.
// rintf == round-half-to-even, matching jnp.round.
// ---------------------------------------------------------------------------
__global__ __launch_bounds__(1024)
void sva_cumsum_kernel(const float* __restrict__ durWS, int* __restrict__ cumc) {
  __shared__ int sd[SEQ];
  const int b = blockIdx.x;
  const int s = threadIdx.x;
  sd[s] = (int)rintf(durWS[(size_t)b * SEQ + s]);
  __syncthreads();
  for (int off = 1; off < SEQ; off <<= 1) {
    const int add = (s >= off) ? sd[s - off] : 0;
    __syncthreads();
    sd[s] += add;
    __syncthreads();
  }
  cumc[(size_t)b * SEQ + s] = min(sd[s], TMAX);
}

// ---------------------------------------------------------------------------
// searchsorted(cumc, t, right) per (b,t) + mel mask.
// ---------------------------------------------------------------------------
__global__ __launch_bounds__(256)
void sva_idx_kernel(const int* __restrict__ cumc,
                    int* __restrict__ idxA,
                    float* __restrict__ maskOut) {
  const int gid = blockIdx.x * blockDim.x + threadIdx.x;  // 0 .. B*T-1
  const int b = gid >> 13;
  const int t = gid & (TMAX - 1);
  const int* c = cumc + (size_t)b * SEQ;
  int lo = 0, hi = SEQ;
  while (lo < hi) {
    const int mid = (lo + hi) >> 1;
    if (c[mid] <= t) lo = mid + 1; else hi = mid;
  }
  maskOut[gid] = (t >= c[SEQ - 1]) ? 1.0f : 0.0f;
  idxA[gid] = min(lo, SEQ - 1);
}

// ---------------------------------------------------------------------------
// expanded[b,t,:] = mask ? 0 : XP[b, idx, :].  256 threads per (b,t) row;
// XP (16MB) is L2-resident so gather reads are cache hits.
// ---------------------------------------------------------------------------
__global__ __launch_bounds__(256)
void sva_gather_kernel(const float* __restrict__ xp,
                       const int* __restrict__ idxA,
                       const float* __restrict__ maskF,
                       float* __restrict__ outExp) {
  const int gid = blockIdx.x;       // 0 .. B*T-1
  const int b = gid >> 13;
  const int row = idxA[gid];
  const float mz = maskF[gid];
  const float v = xp[((size_t)b * SEQ + row) * DIM + threadIdx.x];
  outExp[(size_t)gid * DIM + threadIdx.x] = (mz != 0.0f) ? 0.0f : v;
}

// ---------------------------------------------------------------------------
extern "C" void kernel_launch(void* const* d_in, const int* in_sizes, int n_in,
                              void* d_out, int out_size, void* d_ws, size_t ws_size,
                              hipStream_t stream) {
  (void)in_sizes; (void)n_in; (void)out_size; (void)ws_size;

  // ---- inputs (setup_inputs dict order) ----
  const float*         x     = (const float*)d_in[0];
  const unsigned char* pmask = (const unsigned char*)d_in[1];
  const float*         notep = (const float*)d_in[2];
  const float* ln_g  = (const float*)d_in[3];
  const float* ln_b  = (const float*)d_in[4];
  const float* dp_w1 = (const float*)d_in[5];
  const float* dp_b1 = (const float*)d_in[6];
  const float* dp_w2 = (const float*)d_in[7];
  const float* dp_b2 = (const float*)d_in[8];
  const float* dp_wl = (const float*)d_in[9];
  const float* dp_bl = (const float*)d_in[10];
  const float* pp_w1 = (const float*)d_in[11];
  const float* pp_b1 = (const float*)d_in[12];
  const float* pp_w2 = (const float*)d_in[13];
  const float* pp_b2 = (const float*)d_in[14];
  const float* pp_wl = (const float*)d_in[15];
  const float* pp_bl = (const float*)d_in[16];
  const float* ep_w1 = (const float*)d_in[17];
  const float* ep_b1 = (const float*)d_in[18];
  const float* ep_w2 = (const float*)d_in[19];
  const float* ep_b2 = (const float*)d_in[20];
  const float* ep_wl = (const float*)d_in[21];
  const float* ep_bl = (const float*)d_in[22];
  const float* np_w  = (const float*)d_in[23];
  const float* np_b  = (const float*)d_in[24];
  // d_in[25] = max_length (compile-time TMAX = 8192)

  // ---- outputs (flat concat in return order) ----
  float* out      = (float*)d_out;
  float* out_ld   = out;                                     // [16384]
  float* out_dur  = out + NROWS;                             // [16384]
  float* out_pp   = out + 2 * NROWS;                         // [49152]
  float* out_en   = out + 2 * NROWS + 3 * NROWS;             // [16384]
  float* out_exp  = out + 6 * NROWS;                         // [B*T*D]
  float* out_mask = out_exp + (size_t)BATCH * TMAX * DIM;    // [B*T]

  // ---- workspace layout (~51 MB) ----
  float* XP     = (float*)d_ws;                       // [NROWS*D]
  float* XN     = XP + (size_t)NROWS * DIM;           // [NROWS*D] (LN out / H2 ping-pong)
  float* H1     = XN + (size_t)NROWS * DIM;           // [NROWS*F]
  float* normXN = H1 + (size_t)NROWS * FDIM;          // [NROWS]
  float* normXP = normXN + NROWS;                     // [NROWS]
  float* durWS  = normXP + NROWS;                     // [NROWS]
  unsigned int* red = (unsigned int*)(durWS + NROWS); // [16]
  int* cumc = (int*)(red + 16);                       // [B*S]
  int* idxA = cumc + BATCH * SEQ;                     // [B*T]

  const dim3 gNote(DIM / 16, NROWS / 64);
  const dim3 gConv(FDIM / 16, NROWS / 256);

  sva_init_kernel<<<1, 1, 0, stream>>>(red);

  // x = x + note_pitch @ np_w + np_b
  sva_noteproj_kernel<<<gNote, 32, 0, stream>>>(notep, np_w, np_b, x, XP);

  // layernorm + norms + global min/max
  sva_ln_norm_kernel<<<NROWS, 256, 0, stream>>>(XP, ln_g, ln_b, XN, normXN, normXP, red);

  // duration predictor (convs read XN; H2 reuses XN buffer)
  sva_convgemm_kernel<<<gConv, 128, 0, stream>>>(XN, dp_w1, dp_b1, H1);
  sva_convgemm_kernel<<<gConv, 128, 0, stream>>>(H1, dp_w2, dp_b2, XN);
  sva_dp_final_kernel<<<NROWS, 32, 0, stream>>>(XN, dp_wl, dp_bl, pmask, normXN, red,
                                                out_ld, out_dur, durWS);

  // pitch predictor
  sva_convgemm_kernel<<<gConv, 128, 0, stream>>>(XP, pp_w1, pp_b1, H1);
  sva_convgemm_kernel<<<gConv, 128, 0, stream>>>(H1, pp_w2, pp_b2, XN);
  sva_pp_final_kernel<<<NROWS, 32, 0, stream>>>(XN, pp_wl, pp_bl, pmask, normXP, red, out_pp);

  // energy predictor
  sva_convgemm_kernel<<<gConv, 128, 0, stream>>>(XP, ep_w1, ep_b1, H1);
  sva_convgemm_kernel<<<gConv, 128, 0, stream>>>(H1, ep_w2, ep_b2, XN);
  sva_ep_final_kernel<<<NROWS, 32, 0, stream>>>(XN, ep_wl, ep_bl, pmask, out_en);

  // length regulation
  sva_cumsum_kernel<<<BATCH, SEQ, 0, stream>>>(durWS, cumc);
  sva_idx_kernel<<<(BATCH * TMAX) / 256, 256, 0, stream>>>(cumc, idxA, out_mask);
  sva_gather_kernel<<<BATCH * TMAX, 256, 0, stream>>>(XP, idxA, out_mask, out_exp);
}
